// myLukeSelfAttention_48155173323204
// MI455X (gfx1250) — matrix operand order repaired and checked by reading, not compiled, not run on hardware
//
#include <hip/hip_runtime.h>
#include <hip/hip_bf16.h>
#include <cstdint>

typedef unsigned short u16;
typedef __attribute__((ext_vector_type(16))) __bf16 bf16x16;
typedef __attribute__((ext_vector_type(8)))  float  v8f;

#define NWORD 2048
#define NENT  128
#define NSEQ  2176
#define HID   768
#define NHEAD 12
#define HSZ   64
#define WIN   256
#define NEGV  (-10000.0f)

union Frag16 { uint32_t u[8]; bf16x16 v; };

static __device__ __forceinline__ u16 f2bf(float f) {
    uint32_t u = __float_as_uint(f);
    u += 0x7fffu + ((u >> 16) & 1u);          // round-to-nearest-even
    return (u16)(u >> 16);
}

// A-matrix fragment (16x32 bf16, ISA 7.12.2): lane holds row = lane%16,
// group g = lane>>4; pair i -> k = ((i&4)?16:0) + g*8 + (i&3)*2  (even, 32b loads)
static __device__ __forceinline__ Frag16 load_afrag(const u16* rowk, int g) {
    Frag16 f;
#pragma unroll
    for (int i = 0; i < 8; ++i) {
        int ko = ((i & 4) ? 16 : 0) + g * 8 + (i & 3) * 2;
        f.u[i] = *(const uint32_t*)(rowk + ko);
    }
    return f;
}

// B-matrix fragment (32x16 bf16): lane holds column = lane%16, pair i ->
// k = ((i&4)?16:0) + g*8 + (i&3)*2 + {0,1}; rowk points at row of B^T (i.e. the column), offset k0.
static __device__ __forceinline__ Frag16 load_bfrag(const u16* rowk, int g) {
    Frag16 f;
#pragma unroll
    for (int i = 0; i < 8; ++i) {
        int ko = ((i & 4) ? 16 : 0) + g * 8 + (i & 3) * 2;
        f.u[i] = *(const uint32_t*)(rowk + ko);
    }
    return f;
}

static __device__ __forceinline__ v8f wmma_bf16(const Frag16& a, const Frag16& b, v8f c) {
    return __builtin_amdgcn_wmma_f32_16x16x32_bf16(false, a.v, false, b.v,
                                                   (short)0, c, false, false);
}

// ---------------- fp32 -> bf16 casts ----------------
__global__ __launch_bounds__(256)
void cast_kernel(const float* __restrict__ src, u16* __restrict__ dst, int n) {
    int i = blockIdx.x * 256 + threadIdx.x;
    if (i < n) dst[i] = f2bf(src[i]);
}

// weight (K=in, N=out) row-major -> Wt[n*768 + k] bf16 (transposed)
__global__ __launch_bounds__(256)
void tcast_kernel(const float* __restrict__ src, u16* __restrict__ dst) {
    int i = blockIdx.x * 256 + threadIdx.x;   // i = n*HID + k
    int n = i / HID, k = i - n * HID;
    dst[i] = f2bf(src[k * HID + n]);
}

// ---------------- bf16 GEMM:  C = A(Mx768) * W(768x768) + bias ----------------
// block = 256 threads = 8 waves, block tile 128x64, wave tile 32x32, K-step 32.
// transposed!=0 stores C^T (used for V so attention B-fragments are contiguous).
__global__ __launch_bounds__(256)
void gemm_bf16_kernel(const u16* __restrict__ A, const u16* __restrict__ Wt,
                      const float* __restrict__ bias, u16* __restrict__ C,
                      int out_row_off, int transposed, int ldT)
{
    const int lane = threadIdx.x & 31;
    const int wv   = threadIdx.x >> 5;
    const int g = lane >> 4, l16 = lane & 15;
    const int row0 = blockIdx.y * 128 + (wv & 3) * 32;
    const int col0 = blockIdx.x * 64  + (wv >> 2) * 32;

    v8f acc[2][2] = {};
    const u16* a0p = A  + (size_t)(row0 + l16) * HID;
    const u16* a1p = A  + (size_t)(row0 + 16 + l16) * HID;
    const u16* b0p = Wt + (size_t)(col0 + l16) * HID;
    const u16* b1p = Wt + (size_t)(col0 + 16 + l16) * HID;

    for (int kk = 0; kk < HID; kk += 32) {
        Frag16 a0 = load_afrag(a0p + kk, g);
        Frag16 a1 = load_afrag(a1p + kk, g);
        Frag16 b0 = load_bfrag(b0p + kk, g);
        Frag16 b1 = load_bfrag(b1p + kk, g);
        acc[0][0] = wmma_bf16(a0, b0, acc[0][0]);
        acc[0][1] = wmma_bf16(a0, b1, acc[0][1]);
        acc[1][0] = wmma_bf16(a1, b0, acc[1][0]);
        acc[1][1] = wmma_bf16(a1, b1, acc[1][1]);
    }
#pragma unroll
    for (int jt = 0; jt < 2; ++jt) {
        int c = col0 + jt * 16 + l16;
        float bc = bias[c];
#pragma unroll
        for (int it = 0; it < 2; ++it) {
#pragma unroll
            for (int r = 0; r < 8; ++r) {
                int row = out_row_off + row0 + it * 16 + r + 8 * g;  // C layout: vgpr r, group g
                float val = acc[it][jt][r] + bc;
                if (transposed) C[(size_t)c * ldT + row] = f2bf(val);
                else            C[(size_t)row * HID + c] = f2bf(val);
            }
        }
    }
}

// ---------------- fused banded attention (flash style) ----------------
// grid = (34, 12): 4 waves/block, one 16-query tile per wave, one head per blockIdx.y
__global__ __launch_bounds__(128)
void attn_kernel(const u16* __restrict__ Qw, const u16* __restrict__ Qe,
                 const u16* __restrict__ Kb, const u16* __restrict__ Vt,
                 const float* __restrict__ amask, float* __restrict__ out)
{
    __shared__ u16 pbuf[4][16 * 32];          // wave-private P repack slab (1 KB each)
    const int lane = threadIdx.x & 31;
    const int wv   = threadIdx.x >> 5;
    const int g = lane >> 4, l16 = lane & 15;
    const int qtile = blockIdx.x * 4 + wv;
    const int h = blockIdx.y;
    const int q0 = qtile * 16;
    const bool qword = (q0 < NWORD);
    u16* myp = pbuf[wv];

    float mrun[8], lrun[8];
#pragma unroll
    for (int r = 0; r < 8; ++r) { mrun[r] = -1.0e30f; lrun[r] = 0.0f; }
    v8f o[4] = {};

    for (int phase = 0; phase < 2; ++phase) {
        const u16* Qsrc = (phase == 0) ? Qw : Qe;   // Qw vs word keys, Qe vs entity keys
        int jlo, jhi;
        if (phase == 0) {
            if (qword) {   // band skip: out-of-band scores underflow to exactly 0 in softmax
                jlo = (q0 > WIN) ? ((q0 - WIN) & ~31) : 0;
                int je = q0 + 16 + WIN;
                jhi = (je >= NWORD) ? NWORD : ((je + 31) & ~31);
            } else { jlo = 0; jhi = NWORD; }
        } else { jlo = NWORD; jhi = NSEQ; }
        const bool band = qword && (phase == 0);

        const u16* qrow = Qsrc + (size_t)(q0 + l16) * HID + h * HSZ;
        Frag16 qa0 = load_afrag(qrow, g);           // head-dim 0..31
        Frag16 qa1 = load_afrag(qrow + 32, g);      // head-dim 32..63

        for (int j0 = jlo; j0 < jhi; j0 += 32) {
            v8f s[2];
#pragma unroll
            for (int t = 0; t < 2; ++t) {           // S = Q(16x64) . K^T(64x16) x2 key tiles
                const u16* kbase = Kb + (size_t)(j0 + t * 16 + l16) * HID + h * HSZ;
                Frag16 b0 = load_bfrag(kbase, g);
                Frag16 b1 = load_bfrag(kbase + 32, g);
                v8f acc = {};
                acc = wmma_bf16(qa0, b0, acc);
                acc = wmma_bf16(qa1, b1, acc);
                s[t] = acc;
            }
            float am0 = amask[j0 + l16];
            float am1 = amask[j0 + 16 + l16];
            float fm0 = (am0 != 0.0f) ? NEGV : 0.0f;
            float fm1 = (am1 != 0.0f) ? NEGV : 0.0f;

            float tmax[8];
#pragma unroll
            for (int r = 0; r < 8; ++r) {
                float v0 = s[0][r], v1 = s[1][r];
                if (band) {                         // w2w: float_mask add, band + !=0 gate
                    int m  = q0 + r + 8 * g;
                    int d0 = m - (j0 + l16);
                    int d1 = m - (j0 + 16 + l16);
                    float x0 = v0 + fm0;
                    float x1 = v1 + fm1;
                    v0 = ((d0 <= WIN) && (d0 >= -WIN) && (x0 != 0.0f)) ? x0 : NEGV;
                    v1 = ((d1 <= WIN) && (d1 >= -WIN) && (x1 != 0.0f)) ? x1 : NEGV;
                }
                v0 = v0 * 0.125f + am0;             // /sqrt(64) + attention_mask[j]
                v1 = v1 * 0.125f + am1;
                s[0][r] = v0; s[1][r] = v1;
                tmax[r] = fmaxf(v0, v1);
            }
#pragma unroll
            for (int r = 0; r < 8; ++r) {           // row max across 16-lane group
                float v = tmax[r];
#pragma unroll
                for (int d = 1; d < 16; d <<= 1) v = fmaxf(v, __shfl_xor(v, d, 32));
                tmax[r] = fmaxf(mrun[r], v);        // m_new
            }
            float alpha[8], psum[8];
#pragma unroll
            for (int r = 0; r < 8; ++r) {
                alpha[r] = __expf(mrun[r] - tmax[r]);
                mrun[r]  = tmax[r];
                float p0 = __expf(s[0][r] - tmax[r]);
                float p1 = __expf(s[1][r] - tmax[r]);
                myp[(r + 8 * g) * 32 + l16]      = f2bf(p0);   // C-layout -> LDS row-major
                myp[(r + 8 * g) * 32 + 16 + l16] = f2bf(p1);
                psum[r] = p0 + p1;
            }
            asm volatile("s_wait_dscnt 0" ::: "memory");       // LDS store->load fence (wave-local)
#pragma unroll
            for (int r = 0; r < 8; ++r) {           // row sum across 16-lane group
                float v = psum[r];
#pragma unroll
                for (int d = 1; d < 16; d <<= 1) v += __shfl_xor(v, d, 32);
                lrun[r] = lrun[r] * alpha[r] + v;
            }
            Frag16 pa = load_afrag(myp + l16 * 32, g);   // reload P in A-layout
#pragma unroll
            for (int t4 = 0; t4 < 4; ++t4) {        // O += P(16x32) . V(32x64)
                Frag16 vb = load_bfrag(Vt + (size_t)(h * HSZ + t4 * 16 + l16) * NSEQ + j0, g);
                v8f oo = o[t4];
#pragma unroll
                for (int r = 0; r < 8; ++r) oo[r] *= alpha[r];
                o[t4] = wmma_bf16(pa, vb, oo);
            }
        }
    }

    float inv[8];
#pragma unroll
    for (int r = 0; r < 8; ++r) inv[r] = 1.0f / lrun[r];
#pragma unroll
    for (int t4 = 0; t4 < 4; ++t4) {
#pragma unroll
        for (int r = 0; r < 8; ++r) {
            int row = q0 + r + 8 * g;
            int col = h * HSZ + t4 * 16 + l16;
            out[(size_t)row * HID + col] = o[t4][r] * inv[r];
        }
    }
}

// ---------------- host side ----------------
extern "C" void kernel_launch(void* const* d_in, const int* in_sizes, int n_in,
                              void* d_out, int out_size, void* d_ws, size_t ws_size,
                              hipStream_t stream)
{
    const float* wordh = (const float*)d_in[0];
    const float* enth  = (const float*)d_in[1];
    const float* amask = (const float*)d_in[2];
    // weight/bias order in workspace: k, v, q, w2e, e2w, e2e
    const float* wts[6] = { (const float*)d_in[5],  (const float*)d_in[7],
                            (const float*)d_in[3],  (const float*)d_in[9],
                            (const float*)d_in[11], (const float*)d_in[13] };
    const float* bss[6] = { (const float*)d_in[6],  (const float*)d_in[8],
                            (const float*)d_in[4],  (const float*)d_in[10],
                            (const float*)d_in[12], (const float*)d_in[14] };

    const size_t SH = (size_t)NSEQ * HID;   // 1,671,168 elems
    const size_t HH = (size_t)HID * HID;    //   589,824 elems
    u16* Xc = (u16*)d_ws;                   // concat hidden, bf16      (SH)
    u16* Wt = Xc + SH;                      // 6 transposed weights     (6*HH)
    u16* Kb = Wt + 6 * HH;                  // K rows (key-major)       (SH)
    u16* Vt = Kb + SH;                      // V transposed (col-major) (SH)
    u16* Qw = Vt + SH;                      // queries vs word keys     (SH)
    u16* Qe = Qw + SH;                      // queries vs entity keys   (SH)
    float* out = (float*)d_out;

    cast_kernel<<<(NWORD * HID) / 256, 256, 0, stream>>>(wordh, Xc, NWORD * HID);
    cast_kernel<<<(NENT * HID) / 256, 256, 0, stream>>>(enth, Xc + (size_t)NWORD * HID, NENT * HID);
    for (int i = 0; i < 6; ++i)
        tcast_kernel<<<(HID * HID) / 256, 256, 0, stream>>>(wts[i], Wt + (size_t)i * HH);

    u16 *wt_k = Wt, *wt_v = Wt + HH, *wt_q = Wt + 2 * HH,
        *wt_w2e = Wt + 3 * HH, *wt_e2w = Wt + 4 * HH, *wt_e2e = Wt + 5 * HH;

    gemm_bf16_kernel<<<dim3(12, NSEQ / 128), 256, 0, stream>>>(Xc, wt_k, bss[0], Kb, 0, 0, HID);
    gemm_bf16_kernel<<<dim3(12, NSEQ / 128), 256, 0, stream>>>(Xc, wt_v, bss[1], Vt, 0, 1, NSEQ);
    gemm_bf16_kernel<<<dim3(12, NWORD / 128), 256, 0, stream>>>(Xc, wt_q, bss[2], Qw, 0, 0, HID);
    gemm_bf16_kernel<<<dim3(12, 1), 256, 0, stream>>>(Xc + (size_t)NWORD * HID, wt_e2w, bss[4], Qw, NWORD, 0, HID);
    gemm_bf16_kernel<<<dim3(12, NWORD / 128), 256, 0, stream>>>(Xc, wt_w2e, bss[3], Qe, 0, 0, HID);
    gemm_bf16_kernel<<<dim3(12, 1), 256, 0, stream>>>(Xc + (size_t)NWORD * HID, wt_e2e, bss[5], Qe, NWORD, 0, HID);

    attn_kernel<<<dim3(NSEQ / 16 / 4, NHEAD), 128, 0, stream>>>(Qw, Qe, Kb, Vt, amask, out);
}
